// LinearFullyConnectedGPLayer_40656160423983
// MI455X (gfx1250) — matrix operand order repaired and checked by reading, not compiled, not run on hardware
//
#include <hip/hip_runtime.h>
#include <hip/hip_bf16.h>
#include <math.h>

// ---------------------------------------------------------------------------
// Types for CDNA5 WMMA (wave32): v_wmma_f32_16x16x32_bf16
// ---------------------------------------------------------------------------
typedef __attribute__((ext_vector_type(16))) __bf16          bfrag_t; // A/B frag
typedef __attribute__((ext_vector_type(8)))  float           v8f;     // C/D frag
typedef __attribute__((ext_vector_type(8)))  unsigned short  us8;     // 16B chunk
typedef __attribute__((ext_vector_type(4)))  unsigned int    v4u;     // D# grp0/2/3
typedef __attribute__((ext_vector_type(8)))  unsigned int    v8u;     // D# grp1
union FragU { us8 h[2]; bfrag_t v; };

#define B_TOT 16384
#define C_DIM 512                    // Cin == Ch == Cout
#define PS    (B_TOT * C_DIM)        // blade-plane stride (elements)

__device__ __forceinline__ unsigned short f2bf(float x) {
    unsigned u = __float_as_uint(x);
    unsigned r = u + 0x7FFFu + ((u >> 16) & 1u);   // round-to-nearest-even
    return (unsigned short)(r >> 16);
}

__device__ __forceinline__ unsigned lds_off(const void* p) {
    // flat shared address: low 32 bits are the workgroup-relative LDS offset
    return (unsigned)(unsigned long)p;
}

// ---------------------------------------------------------------------------
// Tensor Data Mover: build D# in SGPRs, issue tensor_load_to_lds (inline asm,
// portable across toolchains). 2-D form (groups 0,1) and 3-D form (0..3).
// data_size=2 bytes; dims/strides in elements. CDNA5 ISA.
// ---------------------------------------------------------------------------
__device__ __forceinline__ void tdm_load_2d(unsigned ldsoff, const void* gaddr,
                                            unsigned tile_d0, unsigned tile_d1,
                                            unsigned tens_d0, unsigned tens_d1,
                                            unsigned long stride0) {
    unsigned long ga = (unsigned long)gaddr;
    v4u g0; v8u g1;
    g0[0] = 1u;                                           // count=1 (valid user D#)
    g0[1] = ldsoff;                                       // lds_addr
    g0[2] = (unsigned)(ga & 0xFFFFFFFFu);                 // global_addr[31:0]
    g0[3] = (unsigned)((ga >> 32) & 0x01FFFFFFu) | (2u << 30); // addr[56:32], type=2
    g1[0] = (1u << 16);                                   // data_size=1 -> 2 bytes
    g1[1] = (tens_d0 & 0xFFFFu) << 16;                    // tensor_dim0[15:0]
    g1[2] = (tens_d0 >> 16) | ((tens_d1 & 0xFFFFu) << 16);// dim0 hi | dim1 lo
    g1[3] = (tens_d1 >> 16) | (tile_d0 << 16);            // dim1 hi | tile_dim0
    g1[4] = tile_d1;                                      // tile_dim1 (tile_dim2=0)
    g1[5] = (unsigned)(stride0 & 0xFFFFFFFFu);            // dim0_stride[31:0]
    g1[6] = (unsigned)((stride0 >> 32) & 0xFFFFu);        // dim0_stride[47:32]
    g1[7] = 0u;
    asm volatile("tensor_load_to_lds %0, %1" :: "s"(g0), "s"(g1) : "memory");
}

__device__ __forceinline__ void tdm_load_3d(unsigned ldsoff, const void* gaddr,
                                            unsigned tile_d0, unsigned tile_d1,
                                            unsigned tile_d2,
                                            unsigned tens_d0, unsigned tens_d1,
                                            unsigned tens_d2,
                                            unsigned long stride0,
                                            unsigned long stride1) {
    unsigned long ga = (unsigned long)gaddr;
    v4u g0; v8u g1; v4u g2; v4u g3;
    g0[0] = 1u;
    g0[1] = ldsoff;
    g0[2] = (unsigned)(ga & 0xFFFFFFFFu);
    g0[3] = (unsigned)((ga >> 32) & 0x01FFFFFFu) | (2u << 30);
    g1[0] = (1u << 16);                                   // data_size = 2 bytes
    g1[1] = (tens_d0 & 0xFFFFu) << 16;
    g1[2] = (tens_d0 >> 16) | ((tens_d1 & 0xFFFFu) << 16);
    g1[3] = (tens_d1 >> 16) | (tile_d0 << 16);
    g1[4] = tile_d1 | (tile_d2 << 16);                    // tile_dim1 | tile_dim2
    g1[5] = (unsigned)(stride0 & 0xFFFFFFFFu);
    g1[6] = (unsigned)((stride0 >> 32) & 0xFFFFu)
          | ((unsigned)(stride1 & 0xFFFFu) << 16);        // dim1_stride[15:0]
    g1[7] = (unsigned)((stride1 >> 16) & 0xFFFFFFFFu);    // dim1_stride[47:16]
    g2[0] = tens_d2;                                      // tensor_dim2
    g2[1] = 0u; g2[2] = 0u; g2[3] = 0u;                   // no dim3 / no iterate
    g3[0] = 0u; g3[1] = 0u; g3[2] = 0u; g3[3] = 0u;
    asm volatile("tensor_load_to_lds %0, %1, %2, %3"
                 :: "s"(g0), "s"(g1), "s"(g2), "s"(g3) : "memory");
}

// ---------------------------------------------------------------------------
// Prep kernels
// ---------------------------------------------------------------------------
__global__ void f32_to_bf16_kernel(const float* __restrict__ src,
                                   unsigned short* __restrict__ dst, int n) {
    int i = blockIdx.x * 256 + threadIdx.x;
    if (i < n) dst[i] = f2bf(src[i]);
}

__global__ void zero_f32_kernel(float* __restrict__ p, int n) {
    int i = blockIdx.x * 256 + threadIdx.x;
    if (i < n) p[i] = 0.0f;
}

// vec1 (B,512,8) f32 -> blade-planar bf16: vec1p[i][b*512+m]
__global__ void transpose_vec1_kernel(const float* __restrict__ v,
                                      unsigned short* __restrict__ p) {
    int idx = blockIdx.x * 256 + threadIdx.x;          // one multivector
    const float4* s = (const float4*)&v[idx * 8];
    float4 a = s[0], b = s[1];
    p[0 * PS + idx] = f2bf(a.x);
    p[1 * PS + idx] = f2bf(a.y);
    p[2 * PS + idx] = f2bf(a.z);
    p[3 * PS + idx] = f2bf(a.w);
    p[4 * PS + idx] = f2bf(b.x);
    p[5 * PS + idx] = f2bf(b.y);
    p[6 * PS + idx] = f2bf(b.z);
    p[7 * PS + idx] = f2bf(b.w);
}

// ---------------------------------------------------------------------------
// Pass 1: L = X_i @ Wl^T + bl, R = X_i @ Wr^T + br (8 blades), Cl(3,0)
// geometric product in epilogue. All staging via TDM (double-buffered X).
// Tile 64 b-rows x 32 ch; 8 waves, wave grid 4(M) x 2(N); 16 acc/wave.
// ---------------------------------------------------------------------------
__global__ __launch_bounds__(256) void gp_pass1(
    const unsigned short* __restrict__ vec1p,  // [8][B*512] bf16 blade-planar
    const unsigned short* __restrict__ Wl,     // (512,512) bf16
    const unsigned short* __restrict__ Wr,     // (512,512) bf16
    const float* __restrict__ bL,
    const float* __restrict__ bR,
    unsigned short* __restrict__ gp)           // [8][B*512] bf16 blade-planar
{
    __shared__ __align__(16) unsigned short Xs[2][8][64][32];  // 2 x 32KB
    __shared__ __align__(16) unsigned short Wls[32][512];      // 32KB
    __shared__ __align__(16) unsigned short Wrs[32][512];      // 32KB

    const int t    = threadIdx.x;
    const int lane = t & 31;
    const int wave = t >> 5;
    const int mt   = wave & 3;
    const int nt   = wave >> 2;
    const int b0   = blockIdx.y * 64;
    const int n0   = blockIdx.x * 32;

    if (wave == 0) {   // TDM issue is once-per-wave (EXEC ignored)
        tdm_load_2d(lds_off(&Wls[0][0]), Wl + n0 * C_DIM, 512, 32, 512, 512, 512);
        tdm_load_2d(lds_off(&Wrs[0][0]), Wr + n0 * C_DIM, 512, 32, 512, 512, 512);
        tdm_load_3d(lds_off(&Xs[0][0][0][0]), vec1p + b0 * C_DIM,
                    32, 64, 8, 512, 1u << 20, 8, 512, (unsigned long)PS);
    }

    v8f accL[8], accR[8];
#pragma unroll
    for (int i = 0; i < 8; ++i)
#pragma unroll
        for (int e = 0; e < 8; ++e) { accL[i][e] = 0.0f; accR[i][e] = 0.0f; }

    const int bn  = nt * 16 + (lane & 15);   // B frag: column N, contiguous K half
    const int bk0 = (lane >> 4) * 16;
    const int ar  = mt * 16 + (lane & 15);   // A frag: row M, contiguous K half
    const int ak0 = (lane >> 4) * 16;

    for (int it = 0; it < 16; ++it) {
        const int kk  = it * 32;
        const int cur = it & 1;
        if (wave == 0) {
            if (it < 15) {
                tdm_load_3d(lds_off(&Xs[cur ^ 1][0][0][0]),
                            vec1p + b0 * C_DIM + kk + 32,
                            32, 64, 8, 512, 1u << 20, 8, 512, (unsigned long)PS);
                __builtin_amdgcn_s_wait_tensorcnt(1);  // current buffers done
            } else {
                __builtin_amdgcn_s_wait_tensorcnt(0);
            }
        }
        __syncthreads();

        FragU fbl, fbr;
        fbl.h[0] = *(const us8*)&Wls[bn][kk + bk0];
        fbl.h[1] = *(const us8*)&Wls[bn][kk + bk0 + 8];
        fbr.h[0] = *(const us8*)&Wrs[bn][kk + bk0];
        fbr.h[1] = *(const us8*)&Wrs[bn][kk + bk0 + 8];
#pragma unroll
        for (int i = 0; i < 8; ++i) {
            FragU fa;
            fa.h[0] = *(const us8*)&Xs[cur][i][ar][ak0];
            fa.h[1] = *(const us8*)&Xs[cur][i][ar][ak0 + 8];
            accL[i] = __builtin_amdgcn_wmma_f32_16x16x32_bf16(
                false, fa.v, false, fbl.v, (short)0, accL[i], false, false);
            accR[i] = __builtin_amdgcn_wmma_f32_16x16x32_bf16(
                false, fa.v, false, fbr.v, (short)0, accR[i], false, false);
        }
        __syncthreads();
    }

    // Epilogue: scalar-blade bias + Cl(3,0) geometric product.
    const int ch    = n0 + nt * 16 + (lane & 15);
    const float bl0 = bL[ch];
    const float br0 = bR[ch];
#pragma unroll
    for (int r = 0; r < 8; ++r) {
        const float l0 = accL[0][r] + bl0, l1 = accL[1][r], l2 = accL[2][r], l3 = accL[3][r];
        const float l4 = accL[4][r],       l5 = accL[5][r], l6 = accL[6][r], l7 = accL[7][r];
        const float r0 = accR[0][r] + br0, r1 = accR[1][r], r2 = accR[2][r], r3 = accR[3][r];
        const float r4 = accR[4][r],       r5 = accR[5][r], r6 = accR[6][r], r7 = accR[7][r];
        const float g0 = l0*r0 + l1*r1 + l2*r2 + l3*r3 - l4*r4 - l5*r5 - l6*r6 - l7*r7;
        const float g1 = l0*r1 + l1*r0 - l2*r4 + l4*r2 - l3*r5 + l5*r3 - l6*r7 - l7*r6;
        const float g2 = l0*r2 + l2*r0 + l1*r4 - l4*r1 - l3*r6 + l6*r3 + l5*r7 + l7*r5;
        const float g3 = l0*r3 + l3*r0 + l1*r5 - l5*r1 + l2*r6 - l6*r2 - l4*r7 - l7*r4;
        const float g4 = l0*r4 + l4*r0 + l1*r2 - l2*r1 + l3*r7 + l7*r3 - l5*r6 + l6*r5;
        const float g5 = l0*r5 + l5*r0 + l1*r3 - l3*r1 - l2*r7 - l7*r2 + l4*r6 - l6*r4;
        const float g6 = l0*r6 + l6*r0 + l2*r3 - l3*r2 + l1*r7 + l7*r1 - l4*r5 + l5*r4;
        const float g7 = l0*r7 + l7*r0 + l1*r6 + l6*r1 - l2*r5 - l5*r2 + l3*r4 + l4*r3;
        const int brow = b0 + mt * 16 + r + ((lane >> 4) << 3);  // D-layout M map
        const int base = brow * C_DIM + ch;
        gp[0 * PS + base] = f2bf(g0);
        gp[1 * PS + base] = f2bf(g1);
        gp[2 * PS + base] = f2bf(g2);
        gp[3 * PS + base] = f2bf(g3);
        gp[4 * PS + base] = f2bf(g4);
        gp[5 * PS + base] = f2bf(g5);
        gp[6 * PS + base] = f2bf(g6);
        gp[7 * PS + base] = f2bf(g7);
    }
}

// ---------------------------------------------------------------------------
// Pass 2: out = gp @ Wo^T + bo (8 blades), f32 out + per-row norm-sum atomics.
// ---------------------------------------------------------------------------
__global__ __launch_bounds__(256) void gp_pass2(
    const unsigned short* __restrict__ gp,     // [8][B*512] bf16 blade-planar
    const unsigned short* __restrict__ Wo,     // (512,512) bf16
    const float* __restrict__ bO,
    float* __restrict__ out,                   // (B,512,8) f32 (= d_out)
    float* __restrict__ normsum)               // (B) f32, pre-zeroed
{
    __shared__ __align__(16) unsigned short Xs[2][8][64][32];
    __shared__ __align__(16) unsigned short Wos[32][512];
    __shared__ float rowsum[64];

    const int t    = threadIdx.x;
    const int lane = t & 31;
    const int wave = t >> 5;
    const int mt   = wave & 3;
    const int nt   = wave >> 2;
    const int b0   = blockIdx.y * 64;
    const int n0   = blockIdx.x * 32;

    if (t < 64) rowsum[t] = 0.0f;

    if (wave == 0) {
        tdm_load_2d(lds_off(&Wos[0][0]), Wo + n0 * C_DIM, 512, 32, 512, 512, 512);
        tdm_load_3d(lds_off(&Xs[0][0][0][0]), gp + b0 * C_DIM,
                    32, 64, 8, 512, 1u << 20, 8, 512, (unsigned long)PS);
    }

    v8f acc[8];
#pragma unroll
    for (int i = 0; i < 8; ++i)
#pragma unroll
        for (int e = 0; e < 8; ++e) acc[i][e] = 0.0f;

    const int bn  = nt * 16 + (lane & 15);
    const int bk0 = (lane >> 4) * 16;
    const int ar  = mt * 16 + (lane & 15);
    const int ak0 = (lane >> 4) * 16;

    for (int it = 0; it < 16; ++it) {
        const int kk  = it * 32;
        const int cur = it & 1;
        if (wave == 0) {
            if (it < 15) {
                tdm_load_3d(lds_off(&Xs[cur ^ 1][0][0][0]),
                            gp + b0 * C_DIM + kk + 32,
                            32, 64, 8, 512, 1u << 20, 8, 512, (unsigned long)PS);
                __builtin_amdgcn_s_wait_tensorcnt(1);
            } else {
                __builtin_amdgcn_s_wait_tensorcnt(0);
            }
        }
        __syncthreads();

        FragU fb;
        fb.h[0] = *(const us8*)&Wos[bn][kk + bk0];
        fb.h[1] = *(const us8*)&Wos[bn][kk + bk0 + 8];
#pragma unroll
        for (int i = 0; i < 8; ++i) {
            FragU fa;
            fa.h[0] = *(const us8*)&Xs[cur][i][ar][ak0];
            fa.h[1] = *(const us8*)&Xs[cur][i][ar][ak0 + 8];
            acc[i] = __builtin_amdgcn_wmma_f32_16x16x32_bf16(
                false, fa.v, false, fb.v, (short)0, acc[i], false, false);
        }
        __syncthreads();
    }

    const int ch   = n0 + nt * 16 + (lane & 15);
    const float bo = bO[ch];
#pragma unroll
    for (int r = 0; r < 8; ++r) {
        const float g0 = acc[0][r] + bo;
        const float g1 = acc[1][r], g2 = acc[2][r], g3 = acc[3][r];
        const float g4 = acc[4][r], g5 = acc[5][r], g6 = acc[6][r], g7 = acc[7][r];
        const int rloc = mt * 16 + r + ((lane >> 4) << 3);
        const int brow = b0 + rloc;
        float4* p = (float4*)&out[(brow * C_DIM + ch) * 8];
        p[0] = make_float4(g0, g1, g2, g3);
        p[1] = make_float4(g4, g5, g6, g7);
        const float nrm = sqrtf(g0*g0 + g1*g1 + g2*g2 + g3*g3 +
                                g4*g4 + g5*g5 + g6*g6 + g7*g7);
        atomicAdd(&rowsum[rloc], nrm);
    }
    __syncthreads();
    if (t < 64) atomicAdd(&normsum[b0 + t], rowsum[t]);
}

// ---------------------------------------------------------------------------
// Pass 3: in-place MVLayerNorm rescale: out *= a_norm[ch] / (mean_norm + eps)
// ---------------------------------------------------------------------------
__global__ __launch_bounds__(256) void gp_pass3(
    float* __restrict__ out,
    const float* __restrict__ normsum,
    const float* __restrict__ a_norm)
{
    int idx = blockIdx.x * 256 + threadIdx.x;   // one multivector per thread
    int ch  = idx & (C_DIM - 1);
    int b   = idx >> 9;
    const float denom = normsum[b] * (1.0f / (float)C_DIM) + 1e-6f;
    const float s = a_norm[ch] / denom;
    float4* p = (float4*)&out[idx * 8];
    float4 x = p[0], y = p[1];
    x.x *= s; x.y *= s; x.z *= s; x.w *= s;
    y.x *= s; y.y *= s; y.z *= s; y.w *= s;
    p[0] = x; p[1] = y;
}

// ---------------------------------------------------------------------------
// Host launcher
// ---------------------------------------------------------------------------
extern "C" void kernel_launch(void* const* d_in, const int* in_sizes, int n_in,
                              void* d_out, int out_size, void* d_ws, size_t ws_size,
                              hipStream_t stream) {
    const float* vec1   = (const float*)d_in[0];
    const float* W_left = (const float*)d_in[1];
    const float* b_left = (const float*)d_in[2];
    const float* W_right= (const float*)d_in[3];
    const float* b_right= (const float*)d_in[4];
    const float* W_out  = (const float*)d_in[5];
    const float* b_out  = (const float*)d_in[6];
    const float* a_norm = (const float*)d_in[7];
    float* out = (float*)d_out;

    // Workspace layout
    const int WN = C_DIM * C_DIM;                        // 262144 per weight
    unsigned short* Wl = (unsigned short*)d_ws;
    unsigned short* Wr = Wl + WN;
    unsigned short* Wo = Wr + WN;
    float* normsum = (float*)(Wo + WN);                  // B floats
    unsigned short* vec1p = (unsigned short*)(normsum + B_TOT); // [8][B*512] bf16
    unsigned short* gpbuf = vec1p + 8 * PS;              // [8][B*512] bf16

    f32_to_bf16_kernel<<<(WN + 255) / 256, 256, 0, stream>>>(W_left,  Wl, WN);
    f32_to_bf16_kernel<<<(WN + 255) / 256, 256, 0, stream>>>(W_right, Wr, WN);
    f32_to_bf16_kernel<<<(WN + 255) / 256, 256, 0, stream>>>(W_out,   Wo, WN);
    zero_f32_kernel<<<B_TOT / 256, 256, 0, stream>>>(normsum, B_TOT);
    transpose_vec1_kernel<<<(B_TOT * C_DIM) / 256, 256, 0, stream>>>(vec1, vec1p);

    // Channel tiles fast (x), b tiles slow (y) -> X slab stays hot in L2.
    dim3 grid(C_DIM / 32, B_TOT / 64);
    gp_pass1<<<grid, 256, 0, stream>>>(vec1p, Wl, Wr, b_left, b_right, gpbuf);
    gp_pass2<<<grid, 256, 0, stream>>>(gpbuf, Wo, b_out, out, normsum);
    gp_pass3<<<(B_TOT * C_DIM) / 256, 256, 0, stream>>>(out, normsum, a_norm);
}
